// DrugRank_90134183674170
// MI455X (gfx1250) — compile-verified, matrix operand order checked by reading, not run on hardware
//
#include <hip/hip_runtime.h>
#include <hip/hip_bf16.h>

typedef __attribute__((ext_vector_type(16))) __bf16        v16bf;
typedef __attribute__((ext_vector_type(8)))  float         v8f;
typedef __attribute__((ext_vector_type(8)))  unsigned int  v8u;
typedef unsigned int u32;

#define N_GRAPHS 8
#define CLL_SIZE 2048
#define N_NODES 160
#define N_EDGES 320
#define EDGE_SIZE 10
#define EH 128
#define NPG 20

// pack 16 f32 (4x float4) -> v16bf fragment
__device__ __forceinline__ v16bf pack16(const float4 c0, const float4 c1,
                                        const float4 c2, const float4 c3) {
  v16bf d;
  d[0]  = (__bf16)c0.x; d[1]  = (__bf16)c0.y; d[2]  = (__bf16)c0.z; d[3]  = (__bf16)c0.w;
  d[4]  = (__bf16)c1.x; d[5]  = (__bf16)c1.y; d[6]  = (__bf16)c1.z; d[7]  = (__bf16)c1.w;
  d[8]  = (__bf16)c2.x; d[9]  = (__bf16)c2.y; d[10] = (__bf16)c2.z; d[11] = (__bf16)c2.w;
  d[12] = (__bf16)c3.x; d[13] = (__bf16)c3.y; d[14] = (__bf16)c3.z; d[15] = (__bf16)c3.w;
  return d;
}

// packed bf16 multiply: one v_pk_mul_bf16 (VOP3P)
__device__ __forceinline__ u32 pkmul_bf16(u32 a, u32 b) {
  u32 d;
  asm("v_pk_mul_bf16 %0, %1, %2" : "=v"(d) : "v"(a), "v"(b));
  return d;
}
// splat f32 -> packed pair of bf16: one v_cvt_pk_bf16_f32
__device__ __forceinline__ u32 splat_bf16(float x) {
  u32 d;
  asm("v_cvt_pk_bf16_f32 %0, %1, %1" : "=v"(d) : "v"(x));
  return d;
}

// element j (0..15) of a fragment maps to k = koff + ((j<8) ? j : j+8)
__device__ __forceinline__ constexpr int kidx(int j) { return (j < 8) ? j : j + 8; }

// ---------------------------------------------------------------------------
// Generic WMMA bf16 GEMM:  C[m,n] = act( sum_k A[row(m),k]*B[k,n] + bias[n] )
//   BKN==0 : B stored [N,K] row-major ; BKN==1 : B stored [K,N] row-major
// ---------------------------------------------------------------------------
template <int BKN>
__global__ __launch_bounds__(32) void gemm_wmma_t(
    const float* __restrict__ A, int lda,
    const int* __restrict__ rowIdx,
    const float* __restrict__ B,
    const float* __restrict__ bias,
    float* __restrict__ C, int ldc,
    int M, int N, int K, int relu)
{
  const int lane = threadIdx.x & 31;
  const int m0 = blockIdx.y * 16;
  const int n0 = blockIdx.x * 16;
  const int l15 = lane & 15;
  const int hi  = (lane >> 4) & 1;
  const int koff = hi ? 8 : 0;

  const int mi = m0 + l15;
  const int ni = n0 + l15;
  const int mc = (mi < M) ? mi : (M - 1);
  const int arow = rowIdx ? rowIdx[mc] : mc;
  const int nc = (ni < N) ? ni : (N - 1);

  const float* __restrict__ Ar = A + (size_t)arow * lda;
  const float* __restrict__ Br = BKN ? (B + nc) : (B + (size_t)nc * K);

  v8f acc = {};

  const float* ap = Ar + koff;
  const float* bp = BKN ? (Br + (size_t)koff * N) : (Br + koff);
  const size_t bstep = BKN ? ((size_t)32 * N) : (size_t)32;

  const int Kmain = K & ~31;
  for (int k0 = 0; k0 < Kmain; k0 += 32) {
    v16bf af, bf;
    #pragma unroll
    for (int j = 0; j < 16; ++j) {
      af[j] = (__bf16)ap[kidx(j)];
      if (BKN) bf[j] = (__bf16)bp[(size_t)kidx(j) * N];
      else     bf[j] = (__bf16)bp[kidx(j)];
    }
    acc = __builtin_amdgcn_wmma_f32_16x16x32_bf16(false, af, false, bf,
                                                  (short)0, acc, false, false);
    ap += 32;
    bp += bstep;
  }
  if (Kmain < K) {  // single guarded tail iteration
    v16bf af, bf;
    #pragma unroll
    for (int j = 0; j < 16; ++j) {
      const int k = koff + kidx(j) + Kmain;
      float a0 = 0.f, b0 = 0.f;
      if (k < K) {
        a0 = Ar[k];
        b0 = BKN ? Br[(size_t)k * N] : Br[k];
      }
      af[j] = (__bf16)a0;
      bf[j] = (__bf16)b0;
    }
    acc = __builtin_amdgcn_wmma_f32_16x16x32_bf16(false, af, false, bf,
                                                  (short)0, acc, false, false);
  }

  if (ni < N) {
    const float bv = bias ? bias[ni] : 0.f;
    #pragma unroll
    for (int r = 0; r < 8; ++r) {
      int mr = m0 + (hi ? 8 : 0) + r;
      if (mr < M) {
        float v = acc[r] + bv;
        if (relu) v = fmaxf(v, 0.f);
        C[(size_t)mr * ldc + ni] = v;
      }
    }
  }
}

// gather-transpose: xeT[i*320 + e] = x[src[e]*din + i];  grid(din), block(320)
__global__ void k_gatherT(const float* __restrict__ x,
                          const int* __restrict__ src,
                          float* __restrict__ xeT, int din) {
  int e = threadIdx.x;
  int i = blockIdx.x;
  xeT[(size_t)i * N_EDGES + e] = x[(size_t)src[e] * din + i];
}

// ---------------------------------------------------------------------------
// NNConv message GEMM:
//   msg[e,o] += sum_i sum_k  xeT[i,e] * h[e,k] * w2[(i*dout+o)*128 + k]
// 5 waves x 4 M-tiles (64 rows/wave): each B fragment (streamed once from HBM
// as b128) feeds 4 back-to-back WMMAs. Loop-invariant h kept in registers as
// bf16; per-row scaling via v_pk_mul_bf16 with lazy v_cvt_pk_bf16_f32 splats.
// x pre-gathered+transposed so the i-loop advances ONE uniform row pointer.
// ---------------------------------------------------------------------------
__global__ __launch_bounds__(160, 1) void gemm_msg_wmma(
    const float* __restrict__ xeT,  // [din, 320]
    int din,
    const float* __restrict__ h,    // [320,128], 16B aligned rows
    const float* __restrict__ w2,   // [din*dout, 128]
    float* __restrict__ msg,        // [320,dout], pre-filled with b2 term
    int dout)
{
  const int lane = threadIdx.x & 31;
  const int wave = threadIdx.x >> 5;        // 0..4
  const int l15 = lane & 15;
  const int hi  = (lane >> 4) & 1;
  const int n0 = blockIdx.x * 16;
  const int ni = n0 + l15;
  const bool nok = (ni < dout);
  const int nc = nok ? ni : (dout - 1);

  const int ebase = wave * 64 + l15;        // edge id, tile t at +16t

  // preload h fragments (bf16) : 4 tiles x 4 k-steps, 8 VGPRs each
  v8u hf[16];
  #pragma unroll
  for (int t = 0; t < 4; ++t) {
    const float4* __restrict__ h4 = (const float4*)(h + (size_t)(ebase + 16 * t) * EH);
    #pragma unroll
    for (int s = 0; s < 4; ++s) {
      const int q = 2 * hi + 8 * s;         // float4 index of chunk0 (k0=32s)
      hf[t * 4 + s] = __builtin_bit_cast(
          v8u, pack16(h4[q], h4[q + 1], h4[q + 4], h4[q + 5]));
    }
  }

  v8f acc[4] = {{}, {}, {}, {}};
  if (nok) {
    #pragma unroll
    for (int t = 0; t < 4; ++t)
      #pragma unroll
      for (int r = 0; r < 8; ++r)
        acc[t][r] = msg[(size_t)(wave * 64 + t * 16 + (hi ? 8 : 0) + r) * dout + ni];
  }

  const float* __restrict__ xrow = xeT + ebase;   // advance by 320 per i
  const float4* __restrict__ w2p = (const float4*)(w2 + (size_t)nc * EH);
  const size_t w2stride = (size_t)dout * (EH / 4);   // float4 per i

  for (int i = 0; i < din; ++i) {
    float xv[4];
    #pragma unroll
    for (int t = 0; t < 4; ++t) xv[t] = xrow[16 * t];
    __builtin_prefetch((const float*)(w2p + w2stride), 0, 1);
    #pragma unroll
    for (int s = 0; s < 4; ++s) {
      const int q = 2 * hi + 8 * s;
      const float4 w0 = w2p[q],     w1 = w2p[q + 1];
      const float4 w4 = w2p[q + 4], w5 = w2p[q + 5];
      const v16bf bfv = pack16(w0, w1, w4, w5);
      #pragma unroll
      for (int t = 0; t < 4; ++t) {
        const u32 xs = splat_bf16(xv[t]);   // lazy: short live range
        v8u au;
        #pragma unroll
        for (int j = 0; j < 8; ++j) au[j] = pkmul_bf16(hf[t * 4 + s][j], xs);
        const v16bf af = __builtin_bit_cast(v16bf, au);
        acc[t] = __builtin_amdgcn_wmma_f32_16x16x32_bf16(
            false, af, false, bfv, (short)0, acc[t], false, false);
      }
    }
    w2p += w2stride;
    xrow += N_EDGES;
  }

  if (nok) {
    #pragma unroll
    for (int t = 0; t < 4; ++t)
      #pragma unroll
      for (int r = 0; r < 8; ++r)
        msg[(size_t)(wave * 64 + t * 16 + (hi ? 8 : 0) + r) * dout + ni] = acc[t][r];
  }
}

// ----------------------- small helper kernels (2-D grids, no int div) ------
__global__ void k_zero(float* p, int n) {
  int i = blockIdx.x * blockDim.x + threadIdx.x;
  if (i < n) p[i] = 0.f;
}

__global__ void k_count(const int* __restrict__ dst, float* __restrict__ cnt) {
  int e = blockIdx.x * blockDim.x + threadIdx.x;
  if (e < N_EDGES) atomicAdd(&cnt[dst[e]], 1.0f);
}

// grid: (ceil(dout/256), N_EDGES)
__global__ void k_scatter(const float* __restrict__ msg,
                          const int* __restrict__ dst,
                          float* __restrict__ agg, int dout) {
  int o = blockIdx.x * blockDim.x + threadIdx.x;
  int e = blockIdx.y;
  if (o < dout)
    atomicAdd(&agg[(size_t)dst[e] * dout + o], msg[(size_t)e * dout + o]);
}

// grid: (ceil(dout/256), N_NODES)
__global__ void k_finalize(const float* __restrict__ agg,
                           const float* __restrict__ cnt,
                           const float* __restrict__ rootmm,
                           float* __restrict__ xout, int dout, int relu) {
  int o = blockIdx.x * blockDim.x + threadIdx.x;
  int n = blockIdx.y;
  if (o < dout) {
    size_t idx = (size_t)n * dout + o;
    float c = fmaxf(cnt[n], 1.0f);
    float v = agg[idx] / c + rootmm[idx];
    if (relu) v = fmaxf(v, 0.f);
    xout[idx] = v;
  }
}

// grid: (ceil(dout/256), N_GRAPHS)
__global__ void k_pool(const float* __restrict__ x, float* __restrict__ zbuf,
                       int dout, int ldz) {
  int o = blockIdx.x * blockDim.x + threadIdx.x;
  int g = blockIdx.y;
  if (o < dout) {
    float s = 0.f;
    for (int t = 0; t < NPG; ++t) s += x[(size_t)(g * NPG + t) * dout + o];
    zbuf[(size_t)g * ldz + o] = s * (1.0f / (float)NPG);
  }
}

// ---------------------------------------------------------------------------
extern "C" void kernel_launch(void* const* d_in, const int* in_sizes, int n_in,
                              void* d_out, int out_size, void* d_ws, size_t ws_size,
                              hipStream_t stream) {
  (void)in_sizes; (void)n_in; (void)out_size; (void)ws_size;

  const float* cll    = (const float*)d_in[0];     // [8, 2048]
  const float* x_mol  = (const float*)d_in[1];     // [160, 75]
  const int*   eidx   = (const int*)d_in[2];       // [2, 320]
  const float* eattr  = (const float*)d_in[3];     // [320, 10]
  const int*   src    = eidx;
  const int*   dst    = eidx + N_EDGES;

  // params pytree order: cll (8 x (w,b)), comb (3 x (w,b)),
  // conv (4 x {b1,b2,bias,root,w1,w2} sorted keys)
  int p = 5;
  const float *cllW[8], *cllBs[8];
  for (int l = 0; l < 8; ++l) { cllW[l] = (const float*)d_in[p++]; cllBs[l] = (const float*)d_in[p++]; }
  const float *cmbW[3], *cmbB[3];
  for (int l = 0; l < 3; ++l) { cmbW[l] = (const float*)d_in[p++]; cmbB[l] = (const float*)d_in[p++]; }
  const float *cv_b1[4], *cv_b2[4], *cv_bias[4], *cv_root[4], *cv_w1[4], *cv_w2[4];
  for (int l = 0; l < 4; ++l) {
    cv_b1[l]   = (const float*)d_in[p++];
    cv_b2[l]   = (const float*)d_in[p++];
    cv_bias[l] = (const float*)d_in[p++];
    cv_root[l] = (const float*)d_in[p++];
    cv_w1[l]   = (const float*)d_in[p++];
    cv_w2[l]   = (const float*)d_in[p++];
  }

  // workspace layout (floats); hE/xeT kept 16B-aligned
  float* ws    = (float*)d_ws;
  float* xA    = ws;                 // 160*1000
  float* xB    = xA + 160 * 1000;    // 160*1000
  float* hE    = xB + 160 * 1000;    // 320*128
  float* msg   = hE + 320 * EH;      // 320*1000
  float* rootm = msg + 320 * 1000;   // 160*1000
  float* agg   = rootm + 160 * 1000; // 160*1000
  float* cnt   = agg + 160 * 1000;   // 256
  float* cA    = cnt + 256;          // 8*2048
  float* cB    = cA + 8 * 2048;      // 8*2048
  float* zbuf  = cB + 8 * 2048;      // 8*512
  float* xeT   = zbuf + 8 * 512;     // 1000*320

  const int ldz = 512;
  dim3 blk32(32);

  // ---------------- CLL MLP tower (M = 8) ----------------
  {
    static const int dims[9] = {2048, 2000, 2000, 2000, 1000, 1000, 900, 800, 400};
    const float* in = cll;
    int K = dims[0];
    float* bufs[2] = {cA, cB};
    for (int l = 0; l < 8; ++l) {
      int N = dims[l + 1];
      float* out = (l == 7) ? (zbuf + 100) : bufs[l & 1];
      int ldc = (l == 7) ? ldz : N;
      int relu = (l < 7) ? 1 : 0;
      dim3 grid((N + 15) / 16, 1);
      hipLaunchKernelGGL((gemm_wmma_t<0>), grid, blk32, 0, stream,
                         in, K, (const int*)nullptr, cllW[l], cllBs[l],
                         out, ldc, N_GRAPHS, N, K, relu);
      in = bufs[l & 1]; K = N;
    }
  }

  // ---------------- NNConv tower ----------------
  {
    static const int dinA[4]  = {75, 1000, 700, 300};
    static const int doutA[4] = {1000, 700, 300, 100};
    const float* xcur = x_mol;
    for (int l = 0; l < 4; ++l) {
      const int din = dinA[l], dout = doutA[l];
      // h = relu(eattr @ w1^T + b1)   [320,128]
      {
        dim3 grid(EH / 16, N_EDGES / 16);
        hipLaunchKernelGGL((gemm_wmma_t<0>), grid, blk32, 0, stream,
                           eattr, EDGE_SIZE, (const int*)nullptr, cv_w1[l], cv_b1[l],
                           hE, EH, N_EDGES, EH, EDGE_SIZE, 1);
      }
      // xeT[i,e] = x[src[e], i]
      hipLaunchKernelGGL(k_gatherT, dim3(din), dim3(N_EDGES), 0, stream,
                         xcur, src, xeT, din);
      // msg init with b2 term: msg[e,o] = sum_i x[src[e],i] * b2[i*dout+o]
      {
        dim3 grid((dout + 15) / 16, N_EDGES / 16);
        hipLaunchKernelGGL((gemm_wmma_t<1>), grid, blk32, 0, stream,
                           xcur, din, src, cv_b2[l], (const float*)nullptr,
                           msg, dout, N_EDGES, dout, din, 0);
      }
      // big message GEMM (accumulates into msg); 5 waves x 4 M-tiles
      {
        dim3 grid((dout + 15) / 16);
        hipLaunchKernelGGL(gemm_msg_wmma, grid, dim3(160), 0, stream,
                           xeT, din, hE, cv_w2[l], msg, dout);
      }
      // root term: rootm = x @ root^T + bias
      {
        dim3 grid((dout + 15) / 16, N_NODES / 16);
        hipLaunchKernelGGL((gemm_wmma_t<0>), grid, blk32, 0, stream,
                           xcur, din, (const int*)nullptr, cv_root[l], cv_bias[l],
                           rootm, dout, N_NODES, dout, din, 0);
      }
      // mean aggregation by dst + finalize (+ relu except last)
      {
        int na = N_NODES * dout;
        hipLaunchKernelGGL(k_zero, dim3((na + 255) / 256), dim3(256), 0, stream, agg, na);
        hipLaunchKernelGGL(k_zero, dim3(1), dim3(256), 0, stream, cnt, 256);
        hipLaunchKernelGGL(k_count, dim3(2), dim3(160), 0, stream, dst, cnt);
        hipLaunchKernelGGL(k_scatter, dim3((dout + 255) / 256, N_EDGES), dim3(256), 0, stream,
                           msg, dst, agg, dout);
        float* xnext = (l & 1) ? xB : xA;
        hipLaunchKernelGGL(k_finalize, dim3((dout + 255) / 256, N_NODES), dim3(256), 0, stream,
                           agg, cnt, rootm, xnext, dout, (l < 3) ? 1 : 0);
        xcur = xnext;
      }
    }
    // global mean pool -> zbuf[:, 0:100]
    hipLaunchKernelGGL(k_pool, dim3(1, N_GRAPHS), dim3(256), 0, stream,
                       xcur, zbuf, 100, ldz);
  }

  // ---------------- combine tower (M = 8): 500 -> 200 -> 100 -> 1 ----------
  {
    dim3 g0((200 + 15) / 16, 1);
    hipLaunchKernelGGL((gemm_wmma_t<0>), g0, blk32, 0, stream,
                       zbuf, ldz, (const int*)nullptr, cmbW[0], cmbB[0],
                       cA, 200, N_GRAPHS, 200, 500, 1);
    dim3 g1((100 + 15) / 16, 1);
    hipLaunchKernelGGL((gemm_wmma_t<0>), g1, blk32, 0, stream,
                       cA, 200, (const int*)nullptr, cmbW[1], cmbB[1],
                       cB, 100, N_GRAPHS, 100, 200, 1);
    dim3 g2(1, 1);
    hipLaunchKernelGGL((gemm_wmma_t<0>), g2, blk32, 0, stream,
                       cB, 100, (const int*)nullptr, cmbW[2], cmbB[2],
                       (float*)d_out, 1, N_GRAPHS, 1, 100, 0);
  }
}
